// FAM_53377853554972
// MI455X (gfx1250) — matrix operand order repaired and checked by reading, not compiled, not run on hardware
//
#include <hip/hip_runtime.h>

// ---------------------------------------------------------------------------
// FAM channel-attention module for MI455X (gfx1250, wave32, WMMA bf16).
// B=4, C=256, N=16384.
//   0) prep: f32->bf16 weights, fold BN scale into Wm, build bias vectors
//   1) gemm_abf: a = Wa x + ba ; b = Wb x + bb ; fT = (Wm' x + shift)^T
//   2) attn_logits: L[c][d] = sum_n a[c][n] b[d][n]   (TDM-staged, dbl-buffered)
//   3) softmax rows of 256, scaled by beta            (bf16)
//   4) out = feature + att' f                         (LDS-free, fT K-contig)
// ---------------------------------------------------------------------------

typedef __bf16 bf16_t;
typedef __attribute__((ext_vector_type(8)))  __bf16 v8bf;
typedef __attribute__((ext_vector_type(16))) __bf16 v16bf;
typedef __attribute__((ext_vector_type(8)))  float  v8f;
typedef __attribute__((ext_vector_type(4)))  float  v4f;
typedef __attribute__((ext_vector_type(4)))  unsigned int u32x4;
typedef __attribute__((ext_vector_type(8)))  int  i32x8;
typedef __attribute__((ext_vector_type(4)))  int  i32x4;

#define C_DIM 256
#define N_DIM 16384
#define B_DIM 4

__device__ __forceinline__ bf16_t f2bf(float f) {
    unsigned u = __builtin_bit_cast(unsigned, f);
    unsigned r = u + 0x7fffu + ((u >> 16) & 1u);   // round-to-nearest-even
    unsigned short s = (unsigned short)(r >> 16);
    return __builtin_bit_cast(bf16_t, s);
}

__device__ __forceinline__ v8f wmma_bf16(v16bf a, v16bf b, v8f c) {
    return __builtin_amdgcn_wmma_f32_16x16x32_bf16(false, a, false, b,
                                                   (short)0, c, false, false);
}

__device__ __forceinline__ v16bf cat16(v8bf lo, v8bf hi) {
    return __builtin_shufflevector(lo, hi, 0,1,2,3,4,5,6,7,8,9,10,11,12,13,14,15);
}

// A fragment: M = lane%16, K = (lane/16)*16 + {0..15}; K contiguous.
__device__ __forceinline__ v16bf load_a_frag(const bf16_t* __restrict__ base,
                                             int row, int k0, int ld, int lane) {
    const bf16_t* p = base + (size_t)(row + (lane & 15)) * ld + k0 + ((lane >> 4) << 4);
    return cat16(*(const v8bf*)(p), *(const v8bf*)(p + 8));
}

// B fragment from K-contiguous rows: N = lane%16, K = (lane/16)*16 + {0..15}.
__device__ __forceinline__ v16bf load_b_frag_rows(const bf16_t* __restrict__ base,
                                                  int row, int k0, int ld, int lane) {
    const bf16_t* p = base + (size_t)(row + (lane & 15)) * ld + k0 + ((lane >> 4) << 4);
    return cat16(*(const v8bf*)(p), *(const v8bf*)(p + 8));
}

// ---------------------------------------------------------------------------
// TDM: 2-D tile DMA (global -> LDS), D# per CDNA5 ISA.
// rows advance by stride_elems in global, packed row-major [rows][cols] in LDS.
// data_size = 2 bytes (bf16). Uniform args only (SGPR descriptors).
// 6-arg builtin form (clang-23 / therock headers).
// ---------------------------------------------------------------------------
__device__ __forceinline__ void tdm_load_2d(const void* gptr, unsigned lds_off,
                                            unsigned rows, unsigned cols,
                                            unsigned stride_elems) {
    unsigned long long ga = (unsigned long long)(size_t)gptr;
    u32x4 g0;
    g0[0] = 1u;                                        // desc count = 1
    g0[1] = lds_off;                                   // LDS byte address
    g0[2] = (unsigned)ga;                              // global_addr[95:64]
    g0[3] = (unsigned)((ga >> 32) & 0x01ffffffu)       // global_addr[120:96]
          | (2u << 30);                                // type = 2 ("image")
    i32x8 g1;
    g1[0] = (int)(1u << 16);                           // data_size=2B, wg_mask=0
    g1[1] = (int)(cols << 16);                         // tensor_dim0[15:0]
    g1[2] = (int)((cols >> 16) | (rows << 16));        // dim0 hi | tensor_dim1 lo
    g1[3] = (int)((rows >> 16) | (cols << 16));        // dim1 hi | tile_dim0
    g1[4] = (int)rows;                                 // tile_dim1 (tile_dim2 = 0)
    g1[5] = (int)stride_elems;                         // tensor_dim0_stride[31:0]
    g1[6] = 0;                                         // stride hi / dim1_stride lo
    g1[7] = 0;
    i32x4 z4 = {0, 0, 0, 0};
    i32x8 z8 = {0, 0, 0, 0, 0, 0, 0, 0};
    __builtin_amdgcn_tensor_load_to_lds(g0, g1, z4, z4, z8, 0);
}

// ---------------------------------------------------------------------------
// Kernel 0: weight conversion + BN fold
// ---------------------------------------------------------------------------
__global__ __launch_bounds__(256) void prep_kernel(
    const float* __restrict__ Wa, const float* __restrict__ ba,
    const float* __restrict__ Wb, const float* __restrict__ bb,
    const float* __restrict__ Wm, const float* __restrict__ gamma,
    const float* __restrict__ bnbeta, const float* __restrict__ mean,
    const float* __restrict__ var,
    bf16_t* __restrict__ Wa_bf, bf16_t* __restrict__ Wb_bf, bf16_t* __restrict__ Wm_bf,
    float* __restrict__ biasA, float* __restrict__ biasB, float* __restrict__ biasF)
{
    int i = blockIdx.x * 256 + threadIdx.x;          // 0 .. 65535
    int o = i >> 8;
    float inv_o = gamma[o] * rsqrtf(var[o] + 1e-5f);
    Wa_bf[i] = f2bf(Wa[i]);
    Wb_bf[i] = f2bf(Wb[i]);
    Wm_bf[i] = f2bf(Wm[i] * inv_o);
    if (i < C_DIM) {
        float inv = gamma[i] * rsqrtf(var[i] + 1e-5f);
        biasA[i] = ba[i];
        biasB[i] = bb[i];
        biasF[i] = bnbeta[i] - mean[i] * inv;
    }
}

// ---------------------------------------------------------------------------
// Kernel 1: fused a/b/f GEMMs. Block tile 128(M) x 64(N), 8 waves of 32x32.
// x tile (f32) is transposed+converted through LDS so B fragments are
// K-contiguous ds_load_b128s; one staging serves all 3 GEMMs.
// f is written TRANSPOSED (fT[n][o]) with packed 16-B stores so that kernel 4
// can consume it K-contiguously without any LDS.
// ---------------------------------------------------------------------------
__global__ __launch_bounds__(256) void gemm_abf_kernel(
    const float* __restrict__ x,                      // [B,C,N]
    const bf16_t* __restrict__ Wa, const bf16_t* __restrict__ Wb,
    const bf16_t* __restrict__ Wm,
    const float* __restrict__ biasA, const float* __restrict__ biasB,
    const float* __restrict__ biasF,
    bf16_t* __restrict__ outA, bf16_t* __restrict__ outB,
    bf16_t* __restrict__ outFT)                       // fT: [B,N,C]
{
    __shared__ bf16_t xs[64][32];                     // [n][k]
    const int n0   = blockIdx.x * 64;
    const int m0   = blockIdx.y * 128;
    const int b    = blockIdx.z;
    const int tid  = threadIdx.x;
    const int lane = tid & 31;
    const int wave = tid >> 5;
    const int wm   = wave & 3;
    const int wn   = wave >> 2;

    const float* xb = x + (size_t)b * C_DIM * N_DIM;
    v8f acc[3][2][2] = {};

    const int stage_k = tid & 31;
    const int stage_n = (tid >> 5) * 8;

    for (int kt = 0; kt < C_DIM; kt += 32) {
        const float* src = xb + (size_t)(kt + stage_k) * N_DIM + n0 + stage_n;
        v4f lo = *(const v4f*)(src);
        v4f hi = *(const v4f*)(src + 4);
        #pragma unroll
        for (int i = 0; i < 4; ++i) xs[stage_n + i][stage_k]     = f2bf(lo[i]);
        #pragma unroll
        for (int i = 0; i < 4; ++i) xs[stage_n + 4 + i][stage_k] = f2bf(hi[i]);
        __syncthreads();

        v16bf bfrag[2];
        #pragma unroll
        for (int j = 0; j < 2; ++j) {
            const bf16_t* p = &xs[wn * 32 + j * 16 + (lane & 15)][(lane >> 4) << 4];
            bfrag[j] = cat16(*(const v8bf*)(p), *(const v8bf*)(p + 8));
        }

        const bf16_t* W[3] = {Wa, Wb, Wm};
        #pragma unroll
        for (int w = 0; w < 3; ++w) {
            #pragma unroll
            for (int i = 0; i < 2; ++i) {
                v16bf afrag = load_a_frag(W[w], m0 + wm * 32 + i * 16, kt, C_DIM, lane);
                #pragma unroll
                for (int j = 0; j < 2; ++j)
                    acc[w][i][j] = wmma_bf16(afrag, bfrag[j], acc[w][i][j]);
            }
        }
        __syncthreads();
    }

    const int voff = (lane >> 4) << 3;
    const int col  = lane & 15;

    // a, b: row-major [c][n]
    bf16_t* outs[2] = {outA, outB};
    const float* biases[2] = {biasA, biasB};
    #pragma unroll
    for (int w = 0; w < 2; ++w)
        #pragma unroll
        for (int i = 0; i < 2; ++i)
            #pragma unroll
            for (int j = 0; j < 2; ++j)
                #pragma unroll
                for (int v = 0; v < 8; ++v) {
                    int row = m0 + wm * 32 + i * 16 + v + voff;
                    int n   = n0 + wn * 32 + j * 16 + col;
                    float val = acc[w][i][j][v] + biases[w][row];
                    outs[w][(size_t)b * C_DIM * N_DIM + (size_t)row * N_DIM + n] = f2bf(val);
                }

    // f: transposed [n][o]; lane's 8 acc elements are contiguous channels.
    #pragma unroll
    for (int i = 0; i < 2; ++i)
        #pragma unroll
        for (int j = 0; j < 2; ++j) {
            int n  = n0 + wn * 32 + j * 16 + col;
            int ob = m0 + wm * 32 + i * 16 + voff;
            v8bf pk;
            #pragma unroll
            for (int v = 0; v < 8; ++v)
                pk[v] = f2bf(acc[2][i][j][v] + biasF[ob + v]);
            *(v8bf*)(outFT + ((size_t)b * N_DIM + n) * C_DIM + ob) = pk;
        }
}

// ---------------------------------------------------------------------------
// Kernel 2: Gram matrix L[c][d] = sum_n a[c][n] b[d][n].
// TDM double-buffered staging: per 64-wide K step, DMA a-tile (128x64) and
// b-tile (64x64) into LDS while WMMA consumes the previous tiles.
// LDS: 2 x (16 KB + 8 KB) = 48 KB dynamic.
// ---------------------------------------------------------------------------
#define K2_STEP   64
#define A_TILE_B  (128 * K2_STEP * 2)                 // 16 KB
#define B_TILE_B  (64  * K2_STEP * 2)                 //  8 KB
#define BUF_B     (A_TILE_B + B_TILE_B)               // 24 KB

__global__ __launch_bounds__(256) void attn_logits_kernel(
    const bf16_t* __restrict__ A, const bf16_t* __restrict__ Bm,
    float* __restrict__ logits)
{
    extern __shared__ char smem[];
    const int d0   = blockIdx.x * 64;
    const int c0   = blockIdx.y * 128;
    const int b    = blockIdx.z;
    const int lane = threadIdx.x & 31;
    const int wave = threadIdx.x >> 5;
    const int wm   = wave & 3;
    const int wn   = wave >> 2;

    const bf16_t* Ab = A  + (size_t)b * C_DIM * N_DIM;
    const bf16_t* Bb = Bm + (size_t)b * C_DIM * N_DIM;
    v8f acc[2][2] = {};

    int cur = 0;
    if (wave == 0) {
        tdm_load_2d(Ab + (size_t)c0 * N_DIM, 0u,        128u, K2_STEP, N_DIM);
        tdm_load_2d(Bb + (size_t)d0 * N_DIM, A_TILE_B,  64u,  K2_STEP, N_DIM);
    }

    for (int kt = 0; kt < N_DIM; kt += K2_STEP) {
        if (wave == 0) __builtin_amdgcn_s_wait_tensorcnt(0);
        __syncthreads();                              // cur buffers ready

        const int nxt = cur ^ 1;
        if (wave == 0 && kt + K2_STEP < N_DIM) {      // prefetch next tiles
            tdm_load_2d(Ab + (size_t)c0 * N_DIM + kt + K2_STEP,
                        (unsigned)(nxt * BUF_B),            128u, K2_STEP, N_DIM);
            tdm_load_2d(Bb + (size_t)d0 * N_DIM + kt + K2_STEP,
                        (unsigned)(nxt * BUF_B + A_TILE_B), 64u,  K2_STEP, N_DIM);
        }

        const bf16_t* as = (const bf16_t*)(smem + cur * BUF_B);
        const bf16_t* bs = (const bf16_t*)(smem + cur * BUF_B + A_TILE_B);
        #pragma unroll
        for (int ks = 0; ks < K2_STEP; ks += 32) {
            v16bf bfrag[2];
            #pragma unroll
            for (int j = 0; j < 2; ++j) {
                const bf16_t* p = bs + (wn * 32 + j * 16 + (lane & 15)) * K2_STEP
                                     + ks + ((lane >> 4) << 4);
                bfrag[j] = cat16(*(const v8bf*)(p), *(const v8bf*)(p + 8));
            }
            #pragma unroll
            for (int i = 0; i < 2; ++i) {
                const bf16_t* p = as + (wm * 32 + i * 16 + (lane & 15)) * K2_STEP
                                     + ks + ((lane >> 4) << 4);
                v16bf afrag = cat16(*(const v8bf*)(p), *(const v8bf*)(p + 8));
                #pragma unroll
                for (int j = 0; j < 2; ++j)
                    acc[i][j] = wmma_bf16(afrag, bfrag[j], acc[i][j]);
            }
        }
        __syncthreads();                              // done reading cur
        cur = nxt;
    }

    const int voff = (lane >> 4) << 3;
    const int col  = lane & 15;
    #pragma unroll
    for (int i = 0; i < 2; ++i)
        #pragma unroll
        for (int j = 0; j < 2; ++j)
            #pragma unroll
            for (int v = 0; v < 8; ++v) {
                int c = c0 + wm * 32 + i * 16 + v + voff;
                int d = d0 + wn * 32 + j * 16 + col;
                logits[(size_t)b * C_DIM * C_DIM + (size_t)c * C_DIM + d] = acc[i][j][v];
            }
}

// ---------------------------------------------------------------------------
// Kernel 3: row softmax (256 wide) scaled by beta; one block per row.
// ---------------------------------------------------------------------------
__global__ __launch_bounds__(256) void softmax_kernel(
    const float* __restrict__ logits, const float* __restrict__ beta,
    bf16_t* __restrict__ att)
{
    const int row  = blockIdx.x;                      // 0 .. B*C-1
    const int tid  = threadIdx.x;
    const int lane = tid & 31;
    const int wave = tid >> 5;
    __shared__ float red[8];
    __shared__ float bcast;

    float v = logits[(size_t)row * C_DIM + tid];

    float m = v;
    #pragma unroll
    for (int o = 16; o; o >>= 1) m = fmaxf(m, __shfl_xor(m, o, 32));
    if (lane == 0) red[wave] = m;
    __syncthreads();
    if (tid == 0) {
        float mm = red[0];
        #pragma unroll
        for (int i = 1; i < 8; ++i) mm = fmaxf(mm, red[i]);
        bcast = mm;
    }
    __syncthreads();
    const float M = bcast;

    float e = __expf(v - M);
    float s = e;
    #pragma unroll
    for (int o = 16; o; o >>= 1) s += __shfl_xor(s, o, 32);
    if (lane == 0) red[wave] = s;
    __syncthreads();
    if (tid == 0) {
        float ss = 0.f;
        #pragma unroll
        for (int i = 0; i < 8; ++i) ss += red[i];
        bcast = beta[0] / ss;                         // fold beta into softmax
    }
    __syncthreads();

    att[(size_t)row * C_DIM + tid] = f2bf(e * bcast);
}

// ---------------------------------------------------------------------------
// Kernel 4: out = feature + att' * f.  LDS-free: att' is K(=d)-contiguous and
// fT[n][d] makes B fragments K-contiguous too; everything streams from L2.
// ---------------------------------------------------------------------------
__global__ __launch_bounds__(256) void out_gemm_kernel(
    const bf16_t* __restrict__ att,                   // [B,C,C]
    const bf16_t* __restrict__ fT,                    // [B,N,C]
    const float* __restrict__ feat, float* __restrict__ out)
{
    const int n0   = blockIdx.x * 64;
    const int m0   = blockIdx.y * 128;
    const int b    = blockIdx.z;
    const int lane = threadIdx.x & 31;
    const int wave = threadIdx.x >> 5;
    const int wm   = wave & 3;
    const int wn   = wave >> 2;

    const bf16_t* attb = att + (size_t)b * C_DIM * C_DIM;
    const bf16_t* fTb  = fT  + (size_t)b * N_DIM * C_DIM;
    v8f acc[2][2] = {};

    for (int kt = 0; kt < C_DIM; kt += 32) {
        v16bf bfrag[2];
        #pragma unroll
        for (int j = 0; j < 2; ++j)
            bfrag[j] = load_b_frag_rows(fTb, n0 + wn * 32 + j * 16, kt, C_DIM, lane);
        #pragma unroll
        for (int i = 0; i < 2; ++i) {
            v16bf afrag = load_a_frag(attb, m0 + wm * 32 + i * 16, kt, C_DIM, lane);
            #pragma unroll
            for (int j = 0; j < 2; ++j)
                acc[i][j] = wmma_bf16(afrag, bfrag[j], acc[i][j]);
        }
    }

    const int voff = (lane >> 4) << 3;
    const int col  = lane & 15;
    #pragma unroll
    for (int i = 0; i < 2; ++i)
        #pragma unroll
        for (int j = 0; j < 2; ++j)
            #pragma unroll
            for (int v = 0; v < 8; ++v) {
                int row = m0 + wm * 32 + i * 16 + v + voff;
                int n   = n0 + wn * 32 + j * 16 + col;
                size_t idx = (size_t)b * C_DIM * N_DIM + (size_t)row * N_DIM + n;
                out[idx] = feat[idx] + acc[i][j][v];
            }
}

// ---------------------------------------------------------------------------
extern "C" void kernel_launch(void* const* d_in, const int* in_sizes, int n_in,
                              void* d_out, int out_size, void* d_ws, size_t ws_size,
                              hipStream_t stream) {
    (void)in_sizes; (void)n_in; (void)out_size; (void)ws_size;
    const float* feature = (const float*)d_in[0];
    const float* Wa      = (const float*)d_in[1];
    const float* ba      = (const float*)d_in[2];
    const float* Wb      = (const float*)d_in[3];
    const float* bb      = (const float*)d_in[4];
    const float* Wm      = (const float*)d_in[5];
    const float* gamma   = (const float*)d_in[6];
    const float* bnbeta  = (const float*)d_in[7];
    const float* mean    = (const float*)d_in[8];
    const float* var     = (const float*)d_in[9];
    const float* beta    = (const float*)d_in[10];
    float* out = (float*)d_out;

    char* ws = (char*)d_ws;
    bf16_t* Wa_bf = (bf16_t*)(ws);
    bf16_t* Wb_bf = (bf16_t*)(ws + (128u << 10));
    bf16_t* Wm_bf = (bf16_t*)(ws + (256u << 10));
    float*  biasA = (float*)(ws + (384u << 10));
    float*  biasB = (float*)(ws + (385u << 10));
    float*  biasF = (float*)(ws + (386u << 10));
    const size_t base = (size_t)512 << 10;
    const size_t matB = (size_t)B_DIM * C_DIM * N_DIM * sizeof(bf16_t);   // 32 MB
    bf16_t* aBuf   = (bf16_t*)(ws + base);
    bf16_t* bBuf   = (bf16_t*)(ws + base + matB);
    bf16_t* fTBuf  = (bf16_t*)(ws + base + 2 * matB);
    float*  logits = (float*)(ws + base + 3 * matB);
    bf16_t* attBuf = (bf16_t*)(ws + base + 3 * matB + ((size_t)1 << 20));

    prep_kernel<<<256, 256, 0, stream>>>(Wa, ba, Wb, bb, Wm, gamma, bnbeta, mean, var,
                                         Wa_bf, Wb_bf, Wm_bf, biasA, biasB, biasF);
    gemm_abf_kernel<<<dim3(N_DIM / 64, C_DIM / 128, B_DIM), 256, 0, stream>>>(
        feature, Wa_bf, Wb_bf, Wm_bf, biasA, biasB, biasF, aBuf, bBuf, fTBuf);
    attn_logits_kernel<<<dim3(C_DIM / 64, C_DIM / 128, B_DIM), 256, 2 * BUF_B, stream>>>(
        aBuf, bBuf, logits);
    softmax_kernel<<<B_DIM * C_DIM, 256, 0, stream>>>(logits, beta, attBuf);
    out_gemm_kernel<<<dim3(N_DIM / 64, C_DIM / 128, B_DIM), 256, 0, stream>>>(
        attBuf, fTBuf, feature, out);
}
